// HybridMemory_59347858096667
// MI455X (gfx1250) — matrix operand order repaired and checked by reading, not compiled, not run on hardware
//
#include <hip/hip_runtime.h>

typedef __attribute__((ext_vector_type(16))) _Float16 v16h;
typedef __attribute__((ext_vector_type(8)))  float    v8f;
typedef __attribute__((ext_vector_type(4)))  _Float16 h4;
typedef __attribute__((ext_vector_type(4)))  unsigned u32x4;
typedef __attribute__((ext_vector_type(8)))  int      i32x8;
typedef __attribute__((ext_vector_type(4)))  int      i32x4;

constexpr int   kB    = 128;     // batch
constexpr int   kN    = 100000;  // memory bank size
constexpr int   kD    = 256;     // feature dim
constexpr int   kC    = 2000;    // num classes
constexpr int   kS    = 751;     // source classes
constexpr float kTemp = 0.05f;
constexpr float kEps  = 1e-6f;

constexpr int DCH    = 16;    // d-chunk for classFeat LDS accumulation (751*16*4B = 48KB LDS)
constexpr int NSPLIT = 4;     // n splits for classFeat accumulation
constexpr int TROWS  = 128;   // feature rows per TDM tile (8 waves x 16 cols)
constexpr int TILES  = 16;    // tiles per block in hard-mining kernel
constexpr int NCH    = TROWS * TILES;                  // 2048 n per block
constexpr int TILE_BYTES = TROWS * kD * 2;             // 64 KB f16 tile
constexpr int TILE_HALVES = TROWS * kD;

// ---- monotonic float<->uint key (for atomic max/min on floats) ----
__device__ __forceinline__ unsigned fkey(float f) {
    unsigned u = __float_as_uint(f);
    return (u & 0x80000000u) ? ~u : (u | 0x80000000u);
}
__device__ __forceinline__ float funkey(unsigned k) {
    unsigned u = (k & 0x80000000u) ? (k ^ 0x80000000u) : ~k;
    return __uint_as_float(u);
}

// ---- WMMA wrapper: D(f32 16x16) = A(f16 16x32) * B(f16 32x16) + C ----
__device__ __forceinline__ v8f wmma16(v16h a, v16h b, v8f c) {
    return __builtin_amdgcn_wmma_f32_16x16x32_f16(
        /*neg_a=*/false, a, /*neg_b=*/false, b,
        /*c_mod=*/(short)0, c, /*reuse_a=*/false, /*reuse_b=*/false);
}

// A-matrix lane packing: lane row m=lane&15,
// half h=lane>=16; VGPR j holds K pair at kb = 2j + 16h (contiguous, matches B).
__device__ __forceinline__ v16h load_A_f32row(const float* row, int kk, int h) {
    v16h a;
#pragma unroll
    for (int j = 0; j < 8; ++j) {
        int kb = 2 * j + 16 * h;
        float2 f = *(const float2*)(row + kk + kb);
        a[2 * j]     = (_Float16)f.x;
        a[2 * j + 1] = (_Float16)f.y;
    }
    return a;
}
// B-matrix lane packing (contiguous k order used here, same as A): lane col n=lane&15, half h:
// VGPR j holds K pair kb = 2j + 16h  ->  halves [kk+16h .. kk+16h+15] CONTIGUOUS.
__device__ __forceinline__ v16h load_B_f32row(const float* row, int kk, int h) {
    v16h b;
#pragma unroll
    for (int j = 0; j < 8; ++j) {
        int kb = 2 * j + 16 * h;
        float2 f = *(const float2*)(row + kk + kb);
        b[2 * j]     = (_Float16)f.x;
        b[2 * j + 1] = (_Float16)f.y;
    }
    return b;
}

// ---- Tensor Data Mover: 2D tile (rows x 256 halves) global->LDS ----
// Builds D# per CDNA5 ISA ch.8 and issues tensor_load_to_lds.
__device__ __forceinline__ void tdm_load_tile_f16(const _Float16* gsrc_rows,
                                                  unsigned lds_byte_addr,
                                                  int rows_in_tensor) {
    unsigned long long ga = (unsigned long long)gsrc_rows;
    int R = rows_in_tensor < 0 ? 0 : rows_in_tensor;

    u32x4 g0;
    g0[0] = 1u;                                   // count=1 (valid), no gather
    g0[1] = lds_byte_addr;                        // lds_addr
    g0[2] = (unsigned)(ga & 0xFFFFFFFFull);       // global_addr[31:0]
    g0[3] = ((unsigned)(ga >> 32) & 0x01FFFFFFu)  // global_addr[56:32]
            | (2u << 30);                         // type=2 ("image")

    i32x8 g1;
    g1[0] = (int)(1u << 16);                      // data_size=1 (2 bytes), mask=0
    g1[1] = (int)(((unsigned)kD & 0xFFFFu) << 16);        // tensor_dim0[15:0]=256
    g1[2] = (int)(((unsigned)R & 0xFFFFu) << 16);         // dim0 hi=0 | tensor_dim1 lo
    g1[3] = (int)((((unsigned)R >> 16) & 0xFFFFu)         // tensor_dim1 hi
            | (((unsigned)kD & 0xFFFFu) << 16));          // tile_dim0=256
    g1[4] = (int)TROWS;                           // tile_dim1=128, tile_dim2=0
    g1[5] = (int)kD;                              // tensor_dim0_stride lo = 256
    g1[6] = 0;                                    // stride hi | dim1_stride lo
    g1[7] = 0;

    i32x4 gz = {0, 0, 0, 0};
#if __clang_major__ >= 23
    i32x8 gz8 = {0, 0, 0, 0, 0, 0, 0, 0};
    __builtin_amdgcn_tensor_load_to_lds(g0, g1, gz, gz, gz8, 0);
#else
    __builtin_amdgcn_tensor_load_to_lds(g0, g1, gz, gz, 0);
#endif
}

// ---------------- init ----------------
__global__ __launch_bounds__(256) void k_init(float* nums, int* is_target,
                                              unsigned* segmax, unsigned* segmin,
                                              float* out) {
    int i = blockIdx.x * 256 + threadIdx.x;
    if (i < kC) { nums[i] = 0.f; is_target[i] = 0; }
    if (i < kC * kB) { segmax[i] = 0u; segmin[i] = 0xFFFFFFFFu; }
    if (i < 2) out[i] = 0.f;
}

// ---------------- class counts + target bitmap ----------------
__global__ __launch_bounds__(256) void k_count(const int* labels, const int* indexes,
                                               float* nums, int* is_target, int* targets) {
    int n = blockIdx.x * 256 + threadIdx.x;
    if (n < kN) atomicAdd(&nums[labels[n]], 1.0f);
    if (n < kB) {
        int t = labels[indexes[n]];
        targets[n] = t;
        is_target[t] = 1;  // benign race
    }
}

// ---------------- features f32 -> f16 (one streaming pass) ----------------
__global__ __launch_bounds__(256) void k_feat16(const float* f, _Float16* o) {
    long i = ((long)blockIdx.x * 256 + threadIdx.x) * 4;  // kN*kD % 4 == 0
    if (i < (long)kN * kD) {
        float4 v = *(const float4*)(f + i);
        h4 r;
        r[0] = (_Float16)v.x; r[1] = (_Float16)v.y;
        r[2] = (_Float16)v.z; r[3] = (_Float16)v.w;
        *(h4*)(o + i) = r;
    }
}

// ---------------- classFeat[c,d] = sum_{label==c, c<S} features[n,d] ----------------
__global__ __launch_bounds__(256) void k_classfeat(const float* features, const int* labels,
                                                   float* partial) {
    __shared__ float acc[kS * DCH];  // 48,064 B
    int dc = blockIdx.x;             // 0..kD/DCH-1
    int p  = blockIdx.y;             // 0..NSPLIT-1
    for (int i = threadIdx.x; i < kS * DCH; i += 256) acc[i] = 0.f;
    __syncthreads();
    int d  = threadIdx.x & (DCH - 1);
    int rl = threadIdx.x >> 4;  // 16 rows in flight
    int n0 = p * (kN / NSPLIT);
    int n1 = n0 + (kN / NSPLIT);
    for (int n = n0 + rl; n < n1; n += 16) {
        int c = labels[n];
        if (c < kS) {
            float v = features[(long)n * kD + dc * DCH + d];
            atomicAdd(&acc[c * DCH + d], v);  // ds_add_f32
        }
    }
    __syncthreads();
    for (int i = threadIdx.x; i < kS * DCH; i += 256) {
        int c = i / DCH, dd = i % DCH;
        partial[((long)p * kS + c) * kD + dc * DCH + dd] = acc[i];
    }
}

__global__ __launch_bounds__(256) void k_reduce(const float* partial, float* classFeat) {
    int i = blockIdx.x * 256 + threadIdx.x;
    if (i < kS * kD) {
        float s = 0.f;
#pragma unroll
        for (int p = 0; p < NSPLIT; ++p) s += partial[(long)p * kS * kD + i];
        classFeat[i] = s;
    }
}

// ---------------- sim/sim2 [S,B] = classFeat @ inputs^T / (TEMP * count) ----------------
__global__ __launch_bounds__(256) void k_sim(const float* classFeat, const float* inputs,
                                             const float* inputs2, const float* nums,
                                             float* sim, float* sim2) {
    int ct   = blockIdx.x;       // c-tile of 16
    int tid  = threadIdx.x;
    int wave = tid >> 5;         // 8 waves -> 8 b-tiles (covers all 128 b)
    int lane = tid & 31;
    int h    = lane >> 4;
    int l16  = lane & 15;

    int c0   = ct * 16;
    int crow = c0 + l16; if (crow >= kS) crow = kS - 1;   // clamp A row (stores guarded)
    const float* arow = classFeat + (long)crow * kD;
    int b0   = wave * 16;
    const float* br1 = inputs  + (long)(b0 + l16) * kD;
    const float* br2 = inputs2 + (long)(b0 + l16) * kD;

    v8f acc1 = {}, acc2 = {};
#pragma unroll
    for (int kk = 0; kk < kD; kk += 32) {
        v16h a  = load_A_f32row(arow, kk, h);
        v16h b1 = load_B_f32row(br1, kk, h);
        v16h b2 = load_B_f32row(br2, kk, h);
        acc1 = wmma16(a, b1, acc1);
        acc2 = wmma16(a, b2, acc2);
    }
    // D layout: value at reg r -> (m = r + 8h, n = lane&15)
#pragma unroll
    for (int r = 0; r < 8; ++r) {
        int c = c0 + r + 8 * h;
        if (c < kS) {
            float nm    = nums[c];
            float dn    = (nm > 0.f) ? nm : 1.0f;
            float scale = 1.0f / (kTemp * dn);
            int   bi    = b0 + l16;
            sim [c * kB + bi] = acc1[r] * scale;
            sim2[c * kB + bi] = acc2[r] * scale;
        }
    }
}

// ---------------- hard mining: TDM-staged f16 logits GEMM + keyed atomics ----------------
__global__ __launch_bounds__(256) void k_hard(const float* inputs, const _Float16* feat16,
                                              const int* labels, const int* is_target,
                                              unsigned* segmax, unsigned* segmin) {
    extern __shared__ _Float16 ldsbuf[];  // 2 x 64KB double-buffered feature tile
    int bblk = blockIdx.x;                // 0..7 (b-slice of 16)
    int tid  = threadIdx.x;
    int wave = tid >> 5;
    int lane = tid & 31;
    int h    = lane >> 4;
    int l16  = lane & 15;
    int nbase = blockIdx.y * NCH;

    // Preload this wave's A operand (16 input rows of its b-slice) into VGPRs: 8 x v16h.
    const float* arow = inputs + (long)(bblk * 16 + l16) * kD;
    v16h aReg[8];
#pragma unroll
    for (int kq = 0; kq < 8; ++kq) aReg[kq] = load_A_f32row(arow, kq * 32, h);

    // Prologue: stage tile 0.
    if (tid < 32) {
        tdm_load_tile_f16(feat16 + (long)nbase * kD, 0u, kN - nbase);
    }

    for (int t = 0; t < TILES; ++t) {
        if (tid < 32) {
            if (t + 1 < TILES) {   // prefetch next tile into other buffer
                int r0 = nbase + (t + 1) * TROWS;
                tdm_load_tile_f16(feat16 + (long)r0 * kD,
                                  (unsigned)(((t + 1) & 1) * TILE_BYTES), kN - r0);
                __builtin_amdgcn_s_wait_tensorcnt(1);  // current tile complete
            } else {
                __builtin_amdgcn_s_wait_tensorcnt(0);
            }
        }
        __syncthreads();
        asm volatile("" ::: "memory");

        // This wave's 16-column tile: LDS rows wave*16 .. wave*16+15.
        const _Float16* brow = ldsbuf + (t & 1) * TILE_HALVES + (wave * 16 + l16) * kD;
        v8f acc = {};
#pragma unroll
        for (int kq = 0; kq < 8; ++kq) {
            v16h b = *(const v16h*)(brow + kq * 32 + 16 * h);  // contiguous ds_load
            acc = wmma16(aReg[kq], b, acc);
        }

        int n = nbase + t * TROWS + wave * 16 + l16;  // this lane's output column
        if (n < kN) {
            int  c   = labels[n];
            bool tgt = (is_target[c] != 0);
            unsigned* pmax = segmax + (long)c * kB + bblk * 16;
            unsigned* pmin = segmin + (long)c * kB + bblk * 16;
#pragma unroll
            for (int r = 0; r < 8; ++r) {
                float v = acc[r] * (1.0f / kTemp);   // logits = dot / TEMP
                unsigned key = fkey(v);
                int b = r + 8 * h;                   // m = r + 8h within the b-slice
                atomicMax(&pmax[b], key);
                if (tgt) atomicMin(&pmin[b], key);
            }
        }
        __syncthreads();  // all reads of buf[t&1] done before it is re-staged
    }
}

// ---------------- finalize: softmax MSE over [B,S] + masked NLL over [B,C] ----------------
__device__ __forceinline__ float blockReduceSum(float v, float* buf) {
    int tid = threadIdx.x;
    buf[tid] = v; __syncthreads();
    for (int s = 128; s > 0; s >>= 1) {
        if (tid < s) buf[tid] += buf[tid + s];
        __syncthreads();
    }
    float r = buf[0]; __syncthreads();
    return r;
}
__device__ __forceinline__ float blockReduceMax(float v, float* buf) {
    int tid = threadIdx.x;
    buf[tid] = v; __syncthreads();
    for (int s = 128; s > 0; s >>= 1) {
        if (tid < s) buf[tid] = fmaxf(buf[tid], buf[tid + s]);
        __syncthreads();
    }
    float r = buf[0]; __syncthreads();
    return r;
}

__global__ __launch_bounds__(256) void k_final(const float* sim, const float* sim2,
                                               const unsigned* segmax, const unsigned* segmin,
                                               const float* nums, const int* targets,
                                               float* out) {
    __shared__ float buf[256];
    int b   = blockIdx.x;
    int tid = threadIdx.x;

    // ---- MSE between softmax(sim[:,b]) and softmax(sim2[:,b]) over S ----
    float m1 = -INFINITY, m2 = -INFINITY;
    for (int s = tid; s < kS; s += 256) {
        m1 = fmaxf(m1, sim [s * kB + b]);
        m2 = fmaxf(m2, sim2[s * kB + b]);
    }
    m1 = blockReduceMax(m1, buf);
    m2 = blockReduceMax(m2, buf);
    float s1 = 0.f, s2 = 0.f;
    for (int s = tid; s < kS; s += 256) {
        s1 += expf(sim [s * kB + b] - m1);
        s2 += expf(sim2[s * kB + b] - m2);
    }
    s1 = blockReduceSum(s1, buf);
    s2 = blockReduceSum(s2, buf);
    float mse = 0.f;
    for (int s = tid; s < kS; s += 256) {
        float d = expf(sim [s * kB + b] - m1) / s1
                - expf(sim2[s * kB + b] - m2) / s2;
        mse += d * d;
    }
    mse = blockReduceSum(mse, buf);
    if (tid == 0) atomicAdd(&out[1], mse / (float)kS);

    // ---- masked-softmax NLL over C (target uses hardest positive = seg_min) ----
    int t = targets[b];
    float sumexp = 0.f;
    for (int c = tid; c < kC; c += 256) {
        if (nums[c] > 0.f) {
            float v = (c == t) ? funkey(segmin[(long)c * kB + b])
                               : funkey(segmax[(long)c * kB + b]);
            sumexp += expf(v);   // reference uses unstabilized exp here
        }
    }
    sumexp = blockReduceSum(sumexp, buf);
    if (tid == 0) {
        float vt = funkey(segmin[(long)t * kB + b]);
        float p  = expf(vt) / (sumexp + kEps);
        atomicAdd(&out[0], -logf(p + kEps) / (float)kB);
    }
}

extern "C" void kernel_launch(void* const* d_in, const int* in_sizes, int n_in,
                              void* d_out, int out_size, void* d_ws, size_t ws_size,
                              hipStream_t stream) {
    const float* inputs   = (const float*)d_in[0];
    const float* inputs2  = (const float*)d_in[1];
    const float* features = (const float*)d_in[2];
    const int*   labels   = (const int*)d_in[3];
    const int*   indexes  = (const int*)d_in[4];
    (void)in_sizes; (void)n_in; (void)out_size; (void)ws_size;
    float* out = (float*)d_out;

    // workspace carve-out (256B aligned)
    char* p = (char*)d_ws;
    auto take = [&](size_t bytes) {
        char* r = p;
        p += (bytes + 255) & ~(size_t)255;
        return r;
    };
    _Float16* feat16    = (_Float16*)take((size_t)kN * kD * 2);           // 51.2 MB
    float*    partial   = (float*)   take((size_t)NSPLIT * kS * kD * 4);  // ~3.1 MB
    float*    classFeat = (float*)   take((size_t)kS * kD * 4);           // ~0.77 MB
    float*    nums      = (float*)   take((size_t)kC * 4);
    int*      is_target = (int*)     take((size_t)kC * 4);
    int*      targets   = (int*)     take((size_t)kB * 4);
    float*    sim       = (float*)   take((size_t)kS * kB * 4);           // ~0.38 MB
    float*    sim2      = (float*)   take((size_t)kS * kB * 4);
    unsigned* segmax    = (unsigned*)take((size_t)kC * kB * 4);           // 1 MB
    unsigned* segmin    = (unsigned*)take((size_t)kC * kB * 4);           // 1 MB

    k_init<<<(kC * kB + 255) / 256, 256, 0, stream>>>(nums, is_target, segmax, segmin, out);
    k_count<<<(kN + 255) / 256, 256, 0, stream>>>(labels, indexes, nums, is_target, targets);
    k_feat16<<<(int)(((long)kN * kD / 4 + 255) / 256), 256, 0, stream>>>(features, feat16);
    k_classfeat<<<dim3(kD / DCH, NSPLIT), 256, 0, stream>>>(features, labels, partial);
    k_reduce<<<(kS * kD + 255) / 256, 256, 0, stream>>>(partial, classFeat);
    k_sim<<<(kS + 15) / 16, 256, 0, stream>>>(classFeat, inputs, inputs2, nums, sim, sim2);
    k_hard<<<dim3(kB / 16, (kN + NCH - 1) / NCH), 256, 2 * TILE_BYTES, stream>>>(
        inputs, feat16, labels, is_target, segmax, segmin);
    k_final<<<kB, 256, 0, stream>>>(sim, sim2, segmax, segmin, nums, targets, out);
}
